// TransformerArchitecture_44220983279858
// MI455X (gfx1250) — matrix operand order repaired and checked by reading, not compiled, not run on hardware
//
#include <hip/hip_runtime.h>
#include <math.h>

#define B_   4
#define S_   1536
#define D_   256
#define H_   8
#define HD_  32
#define DFF_ 1024
#define L_   24

typedef __attribute__((ext_vector_type(16))) __bf16 bf16x16;
typedef __attribute__((ext_vector_type(8)))  __bf16 bf16x8;
typedef __attribute__((ext_vector_type(8)))  float  f32x8;
typedef __attribute__((ext_vector_type(4)))  float  f32x4;

// ---- fragment builders: load f32 (global or LDS), convert to bf16 ---------

// B-fragment (K x N, 16-bit): lane holds column n = n_base + (lane&15),
// K = k0 + 16*(lane>>4) + 0..15 contiguous. p points at W[n][kstart].
__device__ __forceinline__ bf16x16 cvt_b16(const float* p) {
  f32x4 v0 = *(const f32x4*)(p + 0);
  f32x4 v1 = *(const f32x4*)(p + 4);
  f32x4 v2 = *(const f32x4*)(p + 8);
  f32x4 v3 = *(const f32x4*)(p + 12);
  bf16x16 r;
#pragma unroll
  for (int i = 0; i < 4; i++) {
    r[i]      = (__bf16)v0[i];
    r[4 + i]  = (__bf16)v1[i];
    r[8 + i]  = (__bf16)v2[i];
    r[12 + i] = (__bf16)v3[i];
  }
  return r;
}

// A-fragment (M x K=32, 16-bit): lane holds row m = m_base + (lane&15),
// elems 0..15 = K(k0+16*half+0..15) contiguous.
// p points at A[m][k0 + 16*half].
__device__ __forceinline__ bf16x16 cvt_a16(const float* p) {
  f32x4 v0 = *(const f32x4*)(p + 0);
  f32x4 v1 = *(const f32x4*)(p + 4);
  f32x4 v2 = *(const f32x4*)(p + 8);
  f32x4 v3 = *(const f32x4*)(p + 12);
  bf16x16 r;
#pragma unroll
  for (int i = 0; i < 4; i++) {
    r[i]      = (__bf16)v0[i];
    r[4 + i]  = (__bf16)v1[i];
    r[8 + i]  = (__bf16)v2[i];
    r[12 + i] = (__bf16)v3[i];
  }
  return r;
}

__device__ __forceinline__ f32x8 zero8() {
  f32x8 z = {0.f, 0.f, 0.f, 0.f, 0.f, 0.f, 0.f, 0.f};
  return z;
}

// ---- async copy of a 128-row x 32-dword tile from global to LDS -----------
// 1024 16B chunks; 256 threads x 4 chunks. Uses GLOBAL_LOAD_ASYNC_TO_LDS_B128
// (ASYNCcnt). Rows are ldg dwords apart in global; 32 dwords apart in LDS.
__device__ __forceinline__ void async_copy_tile(const float* __restrict__ g,
                                                int ldg, float* lds, int tid) {
#pragma unroll
  for (int c = 0; c < 4; c++) {
    const int ch   = c * 256 + tid;   // 0..1023
    const int row  = ch >> 3;         // 0..127
    const int col4 = ch & 7;          // 16B chunk within row
    const float* gp = g + (size_t)row * ldg + col4 * 4;
    const unsigned loff = (unsigned)(uintptr_t)(lds + row * 32 + col4 * 4);
    asm volatile("global_load_async_to_lds_b128 %0, %1, off"
                 :: "v"(loff), "v"(gp) : "memory");
  }
}

// ---- GEMM: C[M,N] = A[M,K] @ W[N,K]^T + bias, optional exact GELU ---------
// Block: 256 threads = 8 waves as 4(M) x 2(N); block tile 128M x 128N.
// Double-buffered LDS staging via async loads; K-step 32.
// Requires M%128==0, N%128==0, K%32==0.
__global__ void __launch_bounds__(256) gemm_bias_kernel(
    const float* __restrict__ A, const float* __restrict__ W,
    const float* __restrict__ bias, float* __restrict__ C,
    int M, int N, int K, int act) {
  __shared__ __align__(16) float Abuf[2][128 * 32];  // 2 x 16 KB
  __shared__ __align__(16) float Bbuf[2][128 * 32];  // 2 x 16 KB
  const int tid  = threadIdx.x;
  const int lane = tid & 31;
  const int hl   = lane >> 4;
  const int li   = lane & 15;
  const int wv   = tid >> 5;
  const int mw   = wv & 3;    // 0..3 : which 32-row strip
  const int nw   = wv >> 2;   // 0..1 : which 64-col strip

  const int nBlkN = N >> 7;
  const int mBase = (blockIdx.x / nBlkN) << 7;
  const int nBase = (blockIdx.x % nBlkN) << 7;
  const float* Ab = A + (size_t)mBase * K;
  const float* Wb = W + (size_t)nBase * K;

  f32x8 acc[2][4];
#pragma unroll
  for (int s = 0; s < 2; s++)
#pragma unroll
    for (int j = 0; j < 4; j++) acc[s][j] = zero8();

  // prologue: stage k-step 0
  async_copy_tile(Ab, K, Abuf[0], tid);
  async_copy_tile(Wb, K, Bbuf[0], tid);

  const int nsteps = K >> 5;
  for (int step = 0; step < nsteps; step++) {
    const int cur = step & 1;
    if (step + 1 < nsteps) {
      // prefetch next K-step into the other buffer, then release current:
      // async loads complete in order, so waiting <=8 (this thread's 8
      // in-flight prefetches) guarantees the current tile landed.
      async_copy_tile(Ab + (size_t)(step + 1) * 32, K, Abuf[cur ^ 1], tid);
      async_copy_tile(Wb + (size_t)(step + 1) * 32, K, Bbuf[cur ^ 1], tid);
      asm volatile("s_wait_asynccnt 8" ::: "memory");
    } else {
      asm volatile("s_wait_asynccnt 0" ::: "memory");
    }
    __syncthreads();  // whole block's portion of current tile visible

    const float* At = Abuf[cur];
    const float* Bt = Bbuf[cur];
    bf16x16 af[2];
#pragma unroll
    for (int s = 0; s < 2; s++)
      af[s] = cvt_a16(At + (mw * 32 + 16 * s + li) * 32 + 16 * hl);
    bf16x16 bfg[4];
#pragma unroll
    for (int j = 0; j < 4; j++)
      bfg[j] = cvt_b16(Bt + (nw * 64 + 16 * j + li) * 32 + 16 * hl);
#pragma unroll
    for (int s = 0; s < 2; s++)
#pragma unroll
      for (int j = 0; j < 4; j++)
        acc[s][j] = __builtin_amdgcn_wmma_f32_16x16x32_bf16(
            false, af[s], false, bfg[j], (short)0, acc[s][j], false, false);

    __syncthreads();  // all waves done reading before tile is overwritten
  }

  const int m0 = mBase + mw * 32;
  const int n0 = nBase + nw * 64;
#pragma unroll
  for (int j = 0; j < 4; j++) {
    const int col = n0 + 16 * j + li;
    const float bv = bias[col];
#pragma unroll
    for (int s = 0; s < 2; s++) {
#pragma unroll
      for (int e = 0; e < 8; e++) {
        const int row = m0 + 16 * s + 8 * hl + e;
        float v = acc[s][j][e] + bv;
        if (act) v = 0.5f * v * (1.0f + erff(v * 0.70710678f));  // exact GELU
        C[(size_t)row * N + col] = v;
      }
    }
  }
}

// ---- V transpose: Vt[b][h][d][s] = qkv[b][s][2D + h*HD + d] ---------------
__global__ void transpose_v_kernel(const float* __restrict__ qkv,
                                   float* __restrict__ Vt) {
  int idx = blockIdx.x * blockDim.x + threadIdx.x;
  const int total = B_ * H_ * HD_ * S_;
  if (idx >= total) return;
  int s = idx % S_;
  int r = idx / S_;
  int d = r % HD_; r /= HD_;
  int hh = r % H_;
  int b  = r / H_;
  Vt[idx] = qkv[((size_t)b * S_ + s) * (3 * D_) + 2 * D_ + hh * HD_ + d];
}

// ---- flash attention: one wave per (b, h, 16-query tile) ------------------
__global__ void __launch_bounds__(128) attn_kernel(
    const float* __restrict__ qkv, const float* __restrict__ Vt,
    float* __restrict__ O) {
  __shared__ __align__(16) __bf16 Pbuf[4][16 * 32];
  const int lane = threadIdx.x & 31;
  const int hl = lane >> 4;
  const int li = lane & 15;
  const int wv = threadIdx.x >> 5;
  const int wave = blockIdx.x * 4 + wv;
  const int QT = S_ / 16;
  if (wave >= B_ * H_ * QT) return;
  const int qt = wave % QT;
  const int bh = wave / QT;
  const int hh = bh % H_;
  const int b  = bh / H_;
  const int i0 = qt * 16;

  const float* qbase = qkv + (size_t)b * S_ * (3 * D_) + hh * HD_;
  const float* kbase = qbase + D_;
  const float* vtb   = Vt + ((size_t)(b * H_ + hh)) * HD_ * S_;
  __bf16* P = &Pbuf[wv][0];

  // Q fragment is loop-invariant (K dim = head dim = 32).
  bf16x16 qf = cvt_a16(qbase + (size_t)(i0 + li) * (3 * D_) + 16 * hl);

  f32x8 oacc0 = zero8(), oacc1 = zero8();
  float mrow[8], lrow[8];
#pragma unroll
  for (int e = 0; e < 8; e++) { mrow[e] = -1e30f; lrow[e] = 0.f; }

  const int imax = i0 + 15;
  const int jlim_tile = imax - imax % 3 + 2;       // last allowed key for tile
  const int nkt = (jlim_tile + 32) >> 5;           // # of 32-key tiles
  const float scale = 0.17677669529663688f;        // 1/sqrt(32)

  for (int kt = 0; kt < nkt; kt++) {
    const int j0 = kt * 32;
    bf16x16 kf0 = cvt_b16(kbase + (size_t)(j0 + li)      * (3 * D_) + 16 * hl);
    bf16x16 kf1 = cvt_b16(kbase + (size_t)(j0 + 16 + li) * (3 * D_) + 16 * hl);
    f32x8 s0 = zero8(), s1 = zero8();
    s0 = __builtin_amdgcn_wmma_f32_16x16x32_bf16(false, qf, false, kf0, (short)0, s0, false, false);
    s1 = __builtin_amdgcn_wmma_f32_16x16x32_bf16(false, qf, false, kf1, (short)0, s1, false, false);

#pragma unroll
    for (int e = 0; e < 8; e++) {
      const int row = i0 + 8 * hl + e;
      const int lim = row - row % 3 + 2;           // block-causal limit
      float a = ((j0 + li)      <= lim) ? s0[e] * scale : -1e30f;
      float c = ((j0 + 16 + li) <= lim) ? s1[e] * scale : -1e30f;
      // row max across the 16 lanes of this half (wave32 butterfly)
      float t = fmaxf(a, c);
      t = fmaxf(t, __shfl_xor(t, 1, 32));
      t = fmaxf(t, __shfl_xor(t, 2, 32));
      t = fmaxf(t, __shfl_xor(t, 4, 32));
      t = fmaxf(t, __shfl_xor(t, 8, 32));
      const float mn = fmaxf(mrow[e], t);
      const float fr = __expf(mrow[e] - mn);
      mrow[e] = mn;
      const float p0 = __expf(a - mn);
      const float p1 = __expf(c - mn);
      float sm = p0 + p1;
      sm += __shfl_xor(sm, 1, 32);
      sm += __shfl_xor(sm, 2, 32);
      sm += __shfl_xor(sm, 4, 32);
      sm += __shfl_xor(sm, 8, 32);
      lrow[e] = lrow[e] * fr + sm;
      oacc0[e] *= fr;
      oacc1[e] *= fr;
      // stage P to LDS (C-layout -> row-major bf16)
      P[(8 * hl + e) * 32 + li]      = (__bf16)p0;
      P[(8 * hl + e) * 32 + 16 + li] = (__bf16)p1;
    }
    asm volatile("s_wait_dscnt 0" ::: "memory");
    // reload P as an A-fragment (16 queries x 32 keys)
    const __bf16* pp = P + li * 32 + 16 * hl;
    bf16x8 plo = *(const bf16x8*)(pp);
    bf16x8 phi = *(const bf16x8*)(pp + 8);
    bf16x16 pf;
#pragma unroll
    for (int i = 0; i < 8; i++) { pf[i] = plo[i]; pf[8 + i] = phi[i]; }
    // V fragments from transposed V: contiguous 16 keys per lane
    bf16x16 vf0 = cvt_b16(vtb + (size_t)(li)      * S_ + j0 + 16 * hl);
    bf16x16 vf1 = cvt_b16(vtb + (size_t)(16 + li) * S_ + j0 + 16 * hl);
    oacc0 = __builtin_amdgcn_wmma_f32_16x16x32_bf16(false, pf, false, vf0, (short)0, oacc0, false, false);
    oacc1 = __builtin_amdgcn_wmma_f32_16x16x32_bf16(false, pf, false, vf1, (short)0, oacc1, false, false);
  }

#pragma unroll
  for (int e = 0; e < 8; e++) {
    const int row = i0 + 8 * hl + e;
    const float inv = 1.0f / lrow[e];
    O[((size_t)b * S_ + row) * D_ + hh * HD_ + li]      = oacc0[e] * inv;
    O[((size_t)b * S_ + row) * D_ + hh * HD_ + 16 + li] = oacc1[e] * inv;
  }
}

// ---- h = x + pe (pe broadcast over batch) ---------------------------------
__global__ void add_pe_kernel(const float* __restrict__ x,
                              const float* __restrict__ pe,
                              float* __restrict__ h) {
  int idx = blockIdx.x * blockDim.x + threadIdx.x;
  const int total = B_ * S_ * D_;
  if (idx >= total) return;
  h[idx] = x[idx] + pe[idx % (S_ * D_)];
}

// ---- h = LayerNorm(hin + t) * g + b, one block (256 thr) per row ----------
__global__ void __launch_bounds__(256) add_ln_kernel(
    const float* __restrict__ hin, const float* __restrict__ t,
    const float* __restrict__ g, const float* __restrict__ be,
    float* __restrict__ hout) {
  __shared__ float red[256];
  const int row = blockIdx.x;
  const int tid = threadIdx.x;
  const size_t off = (size_t)row * D_ + tid;
  const float y = hin[off] + t[off];
  red[tid] = y;
  __syncthreads();
  for (int s = 128; s > 0; s >>= 1) {
    if (tid < s) red[tid] += red[tid + s];
    __syncthreads();
  }
  const float mean = red[0] * (1.0f / D_);
  __syncthreads();
  const float d = y - mean;
  red[tid] = d * d;
  __syncthreads();
  for (int s = 128; s > 0; s >>= 1) {
    if (tid < s) red[tid] += red[tid + s];
    __syncthreads();
  }
  const float var = red[0] * (1.0f / D_);
  hout[off] = d * rsqrtf(var + 1e-5f) * g[tid] + be[tid];
}

// ---------------------------------------------------------------------------
static inline int gemm_blocks(int M, int N) {
  return (M >> 7) * (N >> 7);  // exact: M,N multiples of 128
}

extern "C" void kernel_launch(void* const* d_in, const int* in_sizes, int n_in,
                              void* d_out, int out_size, void* d_ws, size_t ws_size,
                              hipStream_t stream) {
  (void)in_sizes; (void)n_in; (void)out_size; (void)ws_size;
  const float* x    = (const float*)d_in[0];
  const float* pe   = (const float*)d_in[1];
  const float* Wqkv = (const float*)d_in[2];
  const float* bqkv = (const float*)d_in[3];
  const float* Wo   = (const float*)d_in[4];
  const float* bo   = (const float*)d_in[5];
  const float* W1   = (const float*)d_in[6];
  const float* b1   = (const float*)d_in[7];
  const float* W2   = (const float*)d_in[8];
  const float* b2   = (const float*)d_in[9];
  const float* g1   = (const float*)d_in[10];
  const float* be1  = (const float*)d_in[11];
  const float* g2   = (const float*)d_in[12];
  const float* be2  = (const float*)d_in[13];

  float* h = (float*)d_out;                      // [B,S,D] activation, in-place
  float* ws = (float*)d_ws;
  const int M = B_ * S_;                         // 6144
  float* qkv = ws;                               // M * 3D      = 4.72M f32
  float* Vt  = qkv + (size_t)M * 3 * D_;         // B*H*HD*S    = 1.57M
  float* Ob  = Vt  + (size_t)B_ * H_ * HD_ * S_; // M * D       = 1.57M
  float* tmp = Ob  + (size_t)M * D_;             // M * D       = 1.57M
  float* fb  = tmp + (size_t)M * D_;             // M * DFF     = 6.29M

  const int BSD = M * D_;
  add_pe_kernel<<<(BSD + 255) / 256, 256, 0, stream>>>(x, pe, h);

  const int attn_waves = B_ * H_ * (S_ / 16);    // 3072
  for (int l = 0; l < L_; l++) {
    // qkv = h @ Wqkv^T + bqkv
    gemm_bias_kernel<<<gemm_blocks(M, 3 * D_), 256, 0, stream>>>(
        h, Wqkv + (size_t)l * 3 * D_ * D_, bqkv + (size_t)l * 3 * D_, qkv,
        M, 3 * D_, D_, 0);
    // Vt = transpose(V)
    transpose_v_kernel<<<(B_ * H_ * HD_ * S_ + 255) / 256, 256, 0, stream>>>(qkv, Vt);
    // Ob = softmax(QK^T * scale + mask) V
    attn_kernel<<<attn_waves / 4, 128, 0, stream>>>(qkv, Vt, Ob);
    // tmp = Ob @ Wo^T + bo
    gemm_bias_kernel<<<gemm_blocks(M, D_), 256, 0, stream>>>(
        Ob, Wo + (size_t)l * D_ * D_, bo + (size_t)l * D_, tmp, M, D_, D_, 0);
    // h = LN(h + tmp)
    add_ln_kernel<<<M, 256, 0, stream>>>(h, tmp, g1 + (size_t)l * D_, be1 + (size_t)l * D_, h);
    // fb = gelu(h @ W1^T + b1)
    gemm_bias_kernel<<<gemm_blocks(M, DFF_), 256, 0, stream>>>(
        h, W1 + (size_t)l * DFF_ * D_, b1 + (size_t)l * DFF_, fb, M, DFF_, D_, 1);
    // tmp = fb @ W2^T + b2
    gemm_bias_kernel<<<gemm_blocks(M, D_), 256, 0, stream>>>(
        fb, W2 + (size_t)l * D_ * DFF_, b2 + (size_t)l * D_, tmp, M, D_, DFF_, 0);
    // h = LN(h + tmp)
    add_ln_kernel<<<M, 256, 0, stream>>>(h, tmp, g2 + (size_t)l * D_, be2 + (size_t)l * D_, h);
  }
}
